// MambaSSMModel_23957327577498
// MI455X (gfx1250) — matrix operand order repaired and checked by reading, not compiled, not run on hardware
//
#include <hip/hip_runtime.h>
#include <hip/hip_bf16.h>
#include <math.h>

typedef __bf16 bf16_t;
typedef __attribute__((ext_vector_type(16))) __bf16 v16bf;
typedef __attribute__((ext_vector_type(8)))  __bf16 v8bf;
typedef __attribute__((ext_vector_type(8)))  float  v8f;

// exact parameter types of __builtin_amdgcn_global_load_async_to_lds_b128:
//   (int __vector(4) addrspace(1)*, int __vector(4) addrspace(3)*, imm, imm)
typedef int v4i_t __attribute__((vector_size(16)));
typedef __attribute__((address_space(1))) v4i_t* as1_v4i;
typedef __attribute__((address_space(3))) v4i_t* as3_v4i;

#define DEV __device__ __forceinline__

// ---- model dims ----
#define B_SZ     32
#define L_SEQ    2048
#define NTOK     (B_SZ * L_SEQ)      // 65536 tokens
#define D_MODELc 256
#define D_INNERc 512
#define D_STATEc 16
#define D_CONVc  4
#define N_LAYERSc 4
#define BXS      128                 // padded x_proj output stride

// =====================================================================
// Async global -> LDS staging (CDNA5 ASYNCcnt path), with safe fallback
// =====================================================================
DEV void cp16_g2l(const bf16_t* g, bf16_t* l) {
#if __has_builtin(__builtin_amdgcn_global_load_async_to_lds_b128)
  __builtin_amdgcn_global_load_async_to_lds_b128(
      (as1_v4i)(unsigned long long)g,
      (as3_v4i)(unsigned)(unsigned long long)l, 0, 0);
#else
  *(uint4*)l = *(const uint4*)g;
#endif
}

DEV void wait_async_all() {
#if __has_builtin(__builtin_amdgcn_s_wait_asynccnt)
  __builtin_amdgcn_s_wait_asynccnt(0);
#elif __has_builtin(__builtin_amdgcn_global_load_async_to_lds_b128)
  asm volatile("s_wait_asynccnt 0" ::: "memory");
#endif
}

DEV v16bf combine8(v8bf lo, v8bf hi) {
  v16bf r;
#pragma unroll
  for (int i = 0; i < 8; ++i) { r[i] = lo[i]; r[i + 8] = hi[i]; }
  return r;
}

// =====================================================================
// bf16 WMMA GEMM:  out[M,N] = A[M,K] * W[N,K]^T (+bias) (+epilogue)
//   EPI 0: plain (+bias if given)
//   EPI 1: softplus(x + bias)
//   EPI 2: x + residual[M,N]
// Requirements (guaranteed by caller): M % 64 == 0, N % 128 == 0, K % 64 == 0.
// 256 threads = 8 waves; WG tile 64x128; wave tile 16x64 (1 A-frag x 4 B-frags).
// Double-buffered async global->LDS pipeline (ASYNCcnt).
// =====================================================================
#define BM  64
#define BN  128
#define BK  64
#define LDT 72   // padded LDS row stride (elements) to stagger banks

template <int EPI>
__global__ __launch_bounds__(256) void k_gemm(
    const bf16_t* __restrict__ A, const bf16_t* __restrict__ W,
    const float* __restrict__ bias, const float* __restrict__ res,
    float* __restrict__ out, int M, int N, int K)
{
  __shared__ alignas(16) bf16_t As[2][BM * LDT];   // 2 x  9 KB
  __shared__ alignas(16) bf16_t Bs[2][BN * LDT];   // 2 x 18 KB

  const int tid  = threadIdx.x;
  const int wave = tid >> 5;
  const int lane = tid & 31;
  const int half = lane >> 4;     // wave32 lane[4]: K-half select / M-half select
  const int lr   = lane & 15;

  const int m0 = blockIdx.y * BM;
  const int n0 = blockIdx.x * BN;

  const int tm  = wave >> 1;       // 0..3  (16-row band)
  const int tn0 = (wave & 1) * 4;  // 0 or 4 (four 16-col tiles)

  // straight-line staging: 2 A-chunks + 4 B-chunks of 16B per thread per stage
  auto stage = [&](int k0, int buf) {
#pragma unroll
    for (int i = 0; i < 2; ++i) {
      int c = tid + i * 256, row = c >> 3, cc = (c & 7) * 8;
      cp16_g2l(A + (size_t)(m0 + row) * K + k0 + cc, &As[buf][row * LDT + cc]);
    }
#pragma unroll
    for (int i = 0; i < 4; ++i) {
      int c = tid + i * 256, row = c >> 3, cc = (c & 7) * 8;
      cp16_g2l(W + (size_t)(n0 + row) * K + k0 + cc, &Bs[buf][row * LDT + cc]);
    }
  };

  v8f acc[4] = {{}, {}, {}, {}};

  stage(0, 0);
  const int nk = K / BK;
  for (int kb = 0; kb < nk; ++kb) {
    wait_async_all();
    __syncthreads();                       // current buffer fully staged
    if (kb + 1 < nk) stage((kb + 1) * BK, (kb + 1) & 1);  // overlap next copies
    const bf16_t* Ab = As[kb & 1];
    const bf16_t* Bb = Bs[kb & 1];
#pragma unroll
    for (int kk = 0; kk < BK; kk += 32) {
      // Preload A fragment + all 4 B fragments, then issue WMMAs back-to-back
      // A fragment 16x32: lane=row; elems = K(kk+16h..+15)
      const bf16_t* ar = &Ab[(tm * 16 + lr) * LDT + kk];
      v16bf af = combine8(*(const v8bf*)(ar + half * 16),
                          *(const v8bf*)(ar + half * 16 + 8));
      v16bf bfv[4];
#pragma unroll
      for (int t = 0; t < 4; ++t) {
        // B fragment 32x16: lane=column (W row); elems = K(kk+16h..+15)
        const bf16_t* br = &Bb[((tn0 + t) * 16 + lr) * LDT + kk + half * 16];
        bfv[t] = combine8(*(const v8bf*)br, *(const v8bf*)(br + 8));
      }
#pragma unroll
      for (int t = 0; t < 4; ++t)
        acc[t] = __builtin_amdgcn_wmma_f32_16x16x32_bf16(false, af, false, bfv[t],
                                                         (short)0, acc[t],
                                                         false, false);
    }
    __syncthreads();                       // all waves done reading this buffer
  }

  // epilogue: C/D layout VGPR r -> M = r + 8*half, N = lane&15
  const int rbase = m0 + tm * 16 + half * 8;
#pragma unroll
  for (int t = 0; t < 4; ++t) {
    int col = n0 + (tn0 + t) * 16 + lr;
    float bv = bias ? bias[col] : 0.0f;
#pragma unroll
    for (int r = 0; r < 8; ++r) {
      int gr = rbase + r;
      float v = acc[t][r] + bv;
      if (EPI == 1) v = (v > 20.0f) ? v : log1pf(__expf(v));
      if (EPI == 2) v += res[(size_t)gr * N + col];
      out[(size_t)gr * N + col] = v;
    }
  }
}

// =====================================================================
// Elementwise / small kernels
// =====================================================================
__global__ void k_cvt_bf16(const float* __restrict__ s, bf16_t* __restrict__ d, int n) {
  int i = blockIdx.x * blockDim.x + threadIdx.x;
  if (i < n) d[i] = (bf16_t)s[i];
}

// convert (layers, rows, cols) f32 -> (layers, prows, cols) bf16, zero padding rows
__global__ void k_cvt_pad(const float* __restrict__ s, bf16_t* __restrict__ d,
                          int layers, int rows, int prows, int cols) {
  int i = blockIdx.x * blockDim.x + threadIdx.x;
  if (i >= layers * prows * cols) return;
  int lay = i / (prows * cols), rem = i % (prows * cols);
  int r = rem / cols, c = rem % cols;
  d[i] = (r < rows) ? (bf16_t)s[((size_t)lay * rows + r) * cols + c] : (bf16_t)0.0f;
}

__global__ void k_aneg(const float* __restrict__ alog, float* __restrict__ an, int n) {
  int i = blockIdx.x * blockDim.x + threadIdx.x;
  if (i < n) an[i] = -__expf(alog[i]);
}

__global__ void k_input_proj(const float* __restrict__ x, const float* __restrict__ w,
                             const float* __restrict__ b, float* __restrict__ h,
                             bf16_t* __restrict__ hbf) {
  int idx = blockIdx.x * blockDim.x + threadIdx.x;
  if (idx >= NTOK * D_MODELc) return;
  int tok = idx / D_MODELc, c = idx % D_MODELc;
  const float* xp = x + (size_t)tok * 6;
  const float* wp = w + c * 6;
  float s = b[c];
#pragma unroll
  for (int k = 0; k < 6; ++k) s += xp[k] * wp[k];
  h[idx] = s; hbf[idx] = (bf16_t)s;
}

// causal depthwise conv1d (k=4) + bias + SiLU, on xp half of xz
__global__ void k_conv_silu(const float* __restrict__ xz, const float* __restrict__ cw,
                            const float* __restrict__ cb, float* __restrict__ xc,
                            bf16_t* __restrict__ xcbf) {
  int idx = blockIdx.x * blockDim.x + threadIdx.x;
  if (idx >= NTOK * D_INNERc) return;
  int tok = idx / D_INNERc, d = idx % D_INNERc;
  int l = tok % L_SEQ;
  float s = cb[d];
#pragma unroll
  for (int t = 0; t < D_CONVc; ++t) {
    int ls = l - (D_CONVc - 1) + t;
    if (ls >= 0) s += xz[(size_t)(tok + ls - l) * (2 * D_INNERc) + d] * cw[d * D_CONVc + t];
  }
  s = s / (1.0f + __expf(-s));
  xc[idx] = s; xcbf[idx] = (bf16_t)s;
}

// sequential SSM scan over L; block = (batch b, 128-wide d-chunk)
__global__ __launch_bounds__(128) void k_scan(
    const float* __restrict__ dt, const float* __restrict__ bx,
    const float* __restrict__ xc, const float* __restrict__ xz,
    const float* __restrict__ An, const float* __restrict__ Dp,
    bf16_t* __restrict__ ybf)
{
  const int b = blockIdx.x >> 2;
  const int d = ((blockIdx.x & 3) << 7) + threadIdx.x;
  __shared__ float sBC[2 * D_STATEc];
  float h[D_STATEc], Av[D_STATEc];
#pragma unroll
  for (int n = 0; n < D_STATEc; ++n) { h[n] = 0.0f; Av[n] = An[d * D_STATEc + n]; }
  const float Dv = Dp[d];
  for (int l = 0; l < L_SEQ; ++l) {
    size_t tok = (size_t)b * L_SEQ + l;
    if (threadIdx.x < 2 * D_STATEc)
      sBC[threadIdx.x] = bx[tok * BXS + threadIdx.x];   // padded stride
    __syncthreads();
    float dtv = dt[tok * D_INNERc + d];
    float xv  = xc[tok * D_INNERc + d];
    float zv  = xz[tok * (2 * D_INNERc) + D_INNERc + d];
    float y = 0.0f;
#pragma unroll
    for (int n = 0; n < D_STATEc; ++n) {
      float dA = __expf(Av[n] * dtv);
      h[n] = dA * h[n] + (dtv * sBC[n]) * xv;
      y += h[n] * sBC[D_STATEc + n];
    }
    y += xv * Dv;
    y *= zv / (1.0f + __expf(-zv));      // * silu(z)
    ybf[tok * D_INNERc + d] = (bf16_t)y;
    __syncthreads();
  }
}

// LayerNorm over 256 cols; one wave32 per row, 8 cols/lane, shfl_xor reduce
__global__ __launch_bounds__(256) void k_layernorm256(
    const float* __restrict__ in, const float* __restrict__ g,
    const float* __restrict__ bta, float* __restrict__ out,
    bf16_t* __restrict__ outbf, int rows)
{
  int wave = threadIdx.x >> 5, lane = threadIdx.x & 31;
  int row = blockIdx.x * 8 + wave;
  if (row >= rows) return;
  const float* p = in + (size_t)row * 256;
  float v[8], s = 0.0f;
#pragma unroll
  for (int i = 0; i < 8; ++i) { v[i] = p[lane + i * 32]; s += v[i]; }
#pragma unroll
  for (int m = 16; m >= 1; m >>= 1) s += __shfl_xor(s, m, 32);
  float mean = s * (1.0f / 256.0f);
  float var = 0.0f;
#pragma unroll
  for (int i = 0; i < 8; ++i) { float dd = v[i] - mean; var += dd * dd; }
#pragma unroll
  for (int m = 16; m >= 1; m >>= 1) var += __shfl_xor(var, m, 32);
  float rstd = rsqrtf(var * (1.0f / 256.0f) + 1e-5f);
#pragma unroll
  for (int i = 0; i < 8; ++i) {
    int c = lane + i * 32;
    float o = (v[i] - mean) * rstd * g[c] + bta[c];
    out[(size_t)row * 256 + c] = o;
    if (outbf) outbf[(size_t)row * 256 + c] = (bf16_t)o;
  }
}

__global__ void k_last(const float* __restrict__ h, float* __restrict__ dst) {
  int idx = blockIdx.x * blockDim.x + threadIdx.x;
  if (idx >= B_SZ * D_MODELc) return;
  int r = idx / D_MODELc, c = idx % D_MODELc;
  dst[idx] = h[((size_t)r * L_SEQ + (L_SEQ - 1)) * D_MODELc + c];
}

__global__ void k_concat(const float* __restrict__ tick, const float* __restrict__ sent,
                         const float* __restrict__ ta, float* __restrict__ comb) {
  int idx = blockIdx.x * blockDim.x + threadIdx.x;
  if (idx >= B_SZ * 1036) return;
  int r = idx / 1036, c = idx % 1036;
  float v;
  if (c < 256)       v = tick[r * 256 + c];
  else if (c < 1024) v = sent[r * 768 + (c - 256)];
  else               v = ta[r * 12 + (c - 1024)];
  comb[idx] = v;
}

// small dense layer: out = act(in @ w^T + b); act 1 = exact GELU
__global__ void k_fc(const float* __restrict__ in, const float* __restrict__ w,
                     const float* __restrict__ b, float* __restrict__ out,
                     int rows, int din, int dout, int act) {
  int idx = blockIdx.x * blockDim.x + threadIdx.x;
  if (idx >= rows * dout) return;
  int r = idx / dout, o = idx % dout;
  const float* ip = in + (size_t)r * din;
  const float* wp = w + (size_t)o * din;
  float s = b[o];
  for (int k = 0; k < din; ++k) s += ip[k] * wp[k];
  if (act) s = 0.5f * s * (1.0f + erff(s * 0.7071067811865476f));
  out[idx] = s;
}

// =====================================================================
// Orchestration
// =====================================================================
extern "C" void kernel_launch(void* const* d_in, const int* in_sizes, int n_in,
                              void* d_out, int out_size, void* d_ws, size_t ws_size,
                              hipStream_t stream)
{
  (void)in_sizes; (void)n_in; (void)out_size; (void)ws_size;
  const float* x    = (const float*)d_in[0];
  const float* sent = (const float*)d_in[1];
  const float* ta   = (const float*)d_in[2];
  const float* ipw  = (const float*)d_in[3];
  const float* ipb  = (const float*)d_in[4];
  const float* inw  = (const float*)d_in[5];
  const float* cw   = (const float*)d_in[6];
  const float* cb   = (const float*)d_in[7];
  const float* xpw  = (const float*)d_in[8];
  const float* dtw  = (const float*)d_in[9];
  const float* dtb  = (const float*)d_in[10];
  const float* alog = (const float*)d_in[11];
  const float* Dp   = (const float*)d_in[12];
  const float* outw = (const float*)d_in[13];
  const float* lng  = (const float*)d_in[14];
  const float* lnb  = (const float*)d_in[15];
  const float* eng  = (const float*)d_in[16];
  const float* enb  = (const float*)d_in[17];
  const float* fw1  = (const float*)d_in[18];
  const float* fb1  = (const float*)d_in[19];
  const float* flg  = (const float*)d_in[20];
  const float* flb  = (const float*)d_in[21];
  const float* fw2  = (const float*)d_in[22];
  const float* fb2  = (const float*)d_in[23];
  const float* cl1w = (const float*)d_in[24];
  const float* cl1b = (const float*)d_in[25];
  const float* cl2w = (const float*)d_in[26];
  const float* cl2b = (const float*)d_in[27];
  const float* cl3w = (const float*)d_in[28];
  const float* cl3b = (const float*)d_in[29];

  char* wsb = (char*)d_ws;
  size_t off = 0;
  auto alloc = [&](size_t bytes) -> char* {
    char* p = wsb + off; off += (bytes + 255) & ~(size_t)255; return p;
  };
  float*  h      = (float*) alloc((size_t)NTOK * D_MODELc * 4);
  bf16_t* hbf    = (bf16_t*)alloc((size_t)NTOK * D_MODELc * 2);
  float*  xzbuf  = (float*) alloc((size_t)NTOK * 2 * D_INNERc * 4);
  float*  xcbuf  = (float*) alloc((size_t)NTOK * D_INNERc * 4);
  bf16_t* xcbf   = (bf16_t*)alloc((size_t)NTOK * D_INNERc * 2);
  float*  dtbuf  = (float*) alloc((size_t)NTOK * D_INNERc * 4);
  float*  bxbuf  = (float*) alloc((size_t)NTOK * BXS * 4);
  bf16_t* ybf    = (bf16_t*)alloc((size_t)NTOK * D_INNERc * 2);
  float*  pre    = (float*) alloc((size_t)NTOK * D_MODELc * 4);
  bf16_t* inw_b  = (bf16_t*)alloc((size_t)N_LAYERSc * 1024 * 256 * 2);
  bf16_t* dtw_b  = (bf16_t*)alloc((size_t)N_LAYERSc * 512 * 512 * 2);
  bf16_t* xpw_b  = (bf16_t*)alloc((size_t)N_LAYERSc * BXS * 512 * 2); // zero-padded
  bf16_t* outw_b = (bf16_t*)alloc((size_t)N_LAYERSc * 256 * 512 * 2);
  float*  aneg   = (float*) alloc((size_t)N_LAYERSc * 512 * 16 * 4);
  float*  tickp  = (float*) alloc(32 * 256 * 4);
  float*  tick   = (float*) alloc(32 * 256 * 4);
  float*  comb   = (float*) alloc(32 * 1036 * 4);
  float*  f1     = (float*) alloc(32 * 256 * 4);
  float*  fln    = (float*) alloc(32 * 256 * 4);
  float*  f2     = (float*) alloc(32 * 256 * 4);
  float*  c1     = (float*) alloc(32 * 128 * 4);
  float*  c2     = (float*) alloc(32 * 64 * 4);

  auto cdiv = [](long long a, long long b) { return (unsigned)((a + b - 1) / b); };

  // weight prep (f32 -> bf16; x_proj zero-padded to 128 rows; A = -exp(A_log))
  k_cvt_bf16<<<cdiv(N_LAYERSc*1024*256,256),256,0,stream>>>(inw,  inw_b,  N_LAYERSc*1024*256);
  k_cvt_bf16<<<cdiv(N_LAYERSc*512*512,256),256,0,stream>>>(dtw,  dtw_b,  N_LAYERSc*512*512);
  k_cvt_pad<<<cdiv(N_LAYERSc*BXS*512,256),256,0,stream>>>(xpw, xpw_b,
                                                          N_LAYERSc, 32, BXS, 512);
  k_cvt_bf16<<<cdiv(N_LAYERSc*256*512,256),256,0,stream>>>(outw, outw_b, N_LAYERSc*256*512);
  k_aneg<<<cdiv(N_LAYERSc*512*16,256),256,0,stream>>>(alog, aneg, N_LAYERSc*512*16);

  k_input_proj<<<cdiv((long long)NTOK*D_MODELc,256),256,0,stream>>>(x, ipw, ipb, h, hbf);

  for (int i = 0; i < N_LAYERSc; ++i) {
    { dim3 g(1024/BN, NTOK/BM);   // in_proj
      k_gemm<0><<<g,256,0,stream>>>(hbf, inw_b + (size_t)i*1024*256, nullptr, nullptr,
                                    xzbuf, NTOK, 1024, 256); }
    k_conv_silu<<<cdiv((long long)NTOK*D_INNERc,256),256,0,stream>>>(
        xzbuf, cw + (size_t)i*D_INNERc*D_CONVc, cb + (size_t)i*D_INNERc, xcbuf, xcbf);
    { dim3 g(512/BN, NTOK/BM);    // dt = softplus(xc @ Wdt^T + bdt)
      k_gemm<1><<<g,256,0,stream>>>(xcbf, dtw_b + (size_t)i*512*512,
                                    dtb + (size_t)i*512, nullptr, dtbuf, NTOK, 512, 512); }
    { dim3 g(BXS/BN, NTOK/BM);    // bx = xc @ Wx^T   (N padded 32 -> 128)
      k_gemm<0><<<g,256,0,stream>>>(xcbf, xpw_b + (size_t)i*BXS*512, nullptr, nullptr,
                                    bxbuf, NTOK, BXS, 512); }
    k_scan<<<B_SZ*4,128,0,stream>>>(dtbuf, bxbuf, xcbuf, xzbuf,
                                    aneg + (size_t)i*512*16, Dp + (size_t)i*512, ybf);
    { dim3 g(256/BN, NTOK/BM);    // out_proj + residual
      k_gemm<2><<<g,256,0,stream>>>(ybf, outw_b + (size_t)i*256*512, nullptr,
                                    h, pre, NTOK, 256, 512); }
    k_layernorm256<<<cdiv(NTOK,8),256,0,stream>>>(pre, lng + i*256, lnb + i*256,
                                                  h, hbf, NTOK);
  }

  // head
  k_last<<<cdiv(32*256,256),256,0,stream>>>(h, tickp);
  k_layernorm256<<<cdiv(32,8),256,0,stream>>>(tickp, eng, enb, tick, nullptr, 32);
  k_concat<<<cdiv(32*1036,256),256,0,stream>>>(tick, sent, ta, comb);
  k_fc<<<cdiv(32*256,128),128,0,stream>>>(comb, fw1, fb1, f1, 32, 1036, 256, 1);
  k_layernorm256<<<cdiv(32,8),256,0,stream>>>(f1, flg, flb, fln, nullptr, 32);
  k_fc<<<cdiv(32*256,128),128,0,stream>>>(fln, fw2, fb2, f2, 32, 256, 256, 0);
  k_fc<<<cdiv(32*128,128),128,0,stream>>>(f2, cl1w, cl1b, c1, 32, 256, 128, 1);
  k_fc<<<cdiv(32*64,64),64,0,stream>>>(c1, cl2w, cl2b, c2, 32, 128, 64, 1);
  k_fc<<<cdiv(32*3,96),96,0,stream>>>(c2, cl3w, cl3b, (float*)d_out, 32, 64, 3, 0);
}
